// img_attn_8443905704490
// MI455X (gfx1250) — compile-verified, matrix operand order checked by reading, not run on hardware
//
#include <hip/hip_runtime.h>
#include <hip/hip_bf16.h>

typedef __attribute__((ext_vector_type(16))) __bf16 v16bf;
typedef __attribute__((ext_vector_type(8)))  __bf16 v8bf;
typedef __attribute__((ext_vector_type(8)))  float  v8f;
typedef __attribute__((ext_vector_type(4)))  unsigned int u32x4;
typedef __attribute__((ext_vector_type(8)))  int i32x8;
typedef __attribute__((ext_vector_type(4)))  int i32x4;

#define B_   32
#define NV_  1556
#define NI_  256
#define NT_  1812
#define D_   512
#define ID_  2048

#define KC_   64   // K elements staged per chunk
#define BROW_ 72   // padded LDS row stride in elements (128B row + 16B pad)
#define BBUF_BYTES_ (64 * BROW_ * 2)   // 9216 B per buffer

#ifndef __has_builtin
#define __has_builtin(x) 0
#endif
#if __has_builtin(__builtin_amdgcn_tensor_load_to_lds) && __has_builtin(__builtin_amdgcn_s_wait_tensorcnt)
#define USE_TDM_ 1
#else
#define USE_TDM_ 0
#endif

// ---------------------------------------------------------------------------
// bf16 WMMA GEMM with TDM-staged weight panel in LDS.
// Block = 8 waves = 128 rows x 64 cols. Wave tile 16x64 (4 accumulators).
// Weight panel (64 cols x KC k) staged into LDS once per block per k-chunk by
// the Tensor Data Mover (double buffered). A-fragments stream from global.
// MODE 0: fc    MODE 1: q    MODE 2: k (A row gather)
// MODE 3: proj (+bf16 q addend, remapped)    MODE 4: final (+f32 verts, f32 out)
// ---------------------------------------------------------------------------
template<int MODE>
__global__ __launch_bounds__(256) void wmma_gemm(
    const __bf16* __restrict__ A, const __bf16* __restrict__ W,
    const float*  __restrict__ bias, const __bf16* __restrict__ addB,
    const float*  __restrict__ addF, __bf16* __restrict__ outB,
    float* __restrict__ outF, int K)
{
    __shared__ __bf16 Bsh[2][64 * BROW_];

    const int lane = threadIdx.x & 31;
    const int wv   = threadIdx.x >> 5;
    const int half = lane >> 4;
    const int mr   = lane & 15;
    const long m0  = ((long)blockIdx.x * 8 + wv) * 16;
    const int  n0  = blockIdx.y * 64;

    const long am   = m0 + mr;
    const long arow = (MODE == 2) ? ((am / NV_) * NT_ + (am % NV_)) : am;
    const __bf16* aptr = A + arow * (long)K + half * 8;

    const v8f zero = {0.f,0.f,0.f,0.f,0.f,0.f,0.f,0.f};
    v8f acc[4] = {zero, zero, zero, zero};

    // staging coords for the fallback path
    const int se = threadIdx.x >> 2;          // 0..63  : weight row within panel
    const int sq = (threadIdx.x & 3) * 16;    // 0/16/32/48 : k offset

    auto stage = [&](int buf, int k0) {
#if USE_TDM_
        if (wv == 0) {
            const unsigned long long ga =
                (unsigned long long)(const void*)(W + (long)n0 * K + k0);
            u32x4 g0;
            g0[0] = 1u;                               // count=1, user descriptor
            g0[1] = (unsigned)(buf * BBUF_BYTES_);    // lds_addr (bytes)
            g0[2] = (unsigned)(ga & 0xFFFFFFFFull);   // global_addr[31:0]
            g0[3] = ((unsigned)(ga >> 32) & 0x1FFFFFFu) | (2u << 30); // addr[56:32] | type=2
            i32x8 g1;
            g1[0] = (int)((1u << 16)      // data_size = 2 bytes
                        | (1u << 20)      // pad_enable
                        | (4u << 22)      // pad_interval: 32 dwords (128B)
                        | (3u << 25));    // pad_amount: 4 dwords (16B)
            g1[1] = (int)((unsigned)KC_ << 16);   // tensor_dim0 = KC (low 16)
            g1[2] = (int)(64u << 16);             // tensor_dim1 = 64 rows (low 16)
            g1[3] = (int)((unsigned)KC_ << 16);   // tile_dim0 = KC
            g1[4] = 64;                           // tile_dim1 = 64 rows
            g1[5] = K;                            // tensor_dim0_stride (low 32)
            g1[6] = 0;
            g1[7] = 0;
            i32x4 gz = {0, 0, 0, 0};
            i32x8 gz8 = {0, 0, 0, 0, 0, 0, 0, 0};
            __builtin_amdgcn_tensor_load_to_lds(g0, g1, gz, gz, gz8, 0);
        }
#else
        const __bf16* src = W + (long)(n0 + se) * K + k0 + sq;
        v8bf* dst = (v8bf*)&Bsh[buf][se * BROW_ + sq];
        dst[0] = *(const v8bf*)(src);
        dst[1] = *(const v8bf*)(src + 8);
#endif
    };

    auto publish = [&]() {
#if USE_TDM_
        if (wv == 0) __builtin_amdgcn_s_wait_tensorcnt(0);
#endif
        __syncthreads();
    };

    const int nchunks = K / KC_;
    stage(0, 0);
    for (int c = 0; c < nchunks; ++c) {
        publish();                                   // buf (c&1) is ready
        if (c + 1 < nchunks) stage((c + 1) & 1, (c + 1) * KC_);
        const __bf16* ap = aptr + c * KC_;
        const __bf16* bs = &Bsh[c & 1][half * 8];
#pragma unroll
        for (int kk = 0; kk < KC_; kk += 32) {
            union { v16bf v; v8bf h[2]; } af;
            af.h[0] = *(const v8bf*)(ap + kk);
            af.h[1] = *(const v8bf*)(ap + kk + 16);
            // Batch all B-fragment LDS loads first so the compiler can issue a
            // single dscnt wait and then 4 back-to-back WMMAs (XDL co-exec).
            union { v16bf v; v8bf h[2]; } bfm[4];
#pragma unroll
            for (int t = 0; t < 4; ++t) {
                const __bf16* bp = bs + (t * 16 + mr) * BROW_ + kk;
                bfm[t].h[0] = *(const v8bf*)(bp);
                bfm[t].h[1] = *(const v8bf*)(bp + 16);
            }
#pragma unroll
            for (int t = 0; t < 4; ++t) {
                acc[t] = __builtin_amdgcn_wmma_f32_16x16x32_bf16(
                    false, af.v, false, bfm[t].v, (short)0, acc[t], false, false);
            }
        }
        __syncthreads();                             // done reading buf (c&1)
    }

    // D layout: lane holds col n = mr; vgpr r holds row m0 + r + 8*half
#pragma unroll
    for (int t = 0; t < 4; ++t) {
        const int n = n0 + t*16 + mr;
        const float bv = bias[n];
#pragma unroll
        for (int r = 0; r < 8; ++r) {
            const long mm = m0 + r + 8*half;
            float v = acc[t][r] + bv;
            if (MODE == 0) {
                const long orow = (mm / NI_) * NT_ + NV_ + (mm % NI_);
                outB[orow * D_ + n] = (__bf16)v;
            } else if (MODE == 1 || MODE == 2) {
                outB[mm * D_ + n] = (__bf16)v;
            } else if (MODE == 3) {
                const long qrow = (mm / NV_) * NT_ + (mm % NV_);
                v += (float)addB[qrow * D_ + n];
                outB[mm * D_ + n] = (__bf16)v;
            } else {
                v += addF[mm * D_ + n];
                outF[mm * D_ + n] = v;
            }
        }
    }
}

// ---------------------------------------------------------------------------
// In-place row l2-normalization of a [rows, 512] bf16 matrix; optionally also
// emits logits[r] = dot(row_normalized, w_g) * scale. One 128-thread block/row.
// ---------------------------------------------------------------------------
template<bool LOGIT>
__global__ __launch_bounds__(128) void rownorm_kernel(
    __bf16* __restrict__ X, const float* __restrict__ wg,
    float* __restrict__ logits, float scale)
{
    __shared__ float red[128];
    const long r = blockIdx.x;
    const int  t = threadIdx.x;
    __bf16* row = X + r * D_;
    float v[4];
    float ss = 0.f;
#pragma unroll
    for (int i = 0; i < 4; ++i) { v[i] = (float)row[t*4+i]; ss += v[i]*v[i]; }
    red[t] = ss; __syncthreads();
    for (int s = 64; s > 0; s >>= 1) { if (t < s) red[t] += red[t+s]; __syncthreads(); }
    const float inv = 1.f / fmaxf(sqrtf(red[0]), 1e-12f);
    __syncthreads();
    float dot = 0.f;
#pragma unroll
    for (int i = 0; i < 4; ++i) {
        const float nv = v[i] * inv;
        row[t*4+i] = (__bf16)nv;
        if (LOGIT) dot += nv * wg[t*4+i];
    }
    if (LOGIT) {
        red[t] = dot; __syncthreads();
        for (int s = 64; s > 0; s >>= 1) { if (t < s) red[t] += red[t+s]; __syncthreads(); }
        if (t == 0) logits[r] = red[0] * scale;
    }
}

// ---------------------------------------------------------------------------
// Per-batch softmax over NT_ logits + global query g[b,d] = sum_n p_n * q[b,n,d]
// ---------------------------------------------------------------------------
__global__ __launch_bounds__(256) void softmax_g_kernel(
    const float* __restrict__ logits, const __bf16* __restrict__ qb,
    float* __restrict__ g)
{
    __shared__ float p[NT_];
    __shared__ float red[256];
    const int b = blockIdx.x, t = threadIdx.x;
    const float* lg = logits + (long)b * NT_;
    float mx = -1e30f;
    for (int i = t; i < NT_; i += 256) mx = fmaxf(mx, lg[i]);
    red[t] = mx; __syncthreads();
    for (int s = 128; s > 0; s >>= 1) { if (t < s) red[t] = fmaxf(red[t], red[t+s]); __syncthreads(); }
    mx = red[0]; __syncthreads();
    float sum = 0.f;
    for (int i = t; i < NT_; i += 256) { const float e = __expf(lg[i]-mx); p[i] = e; sum += e; }
    red[t] = sum; __syncthreads();
    for (int s = 128; s > 0; s >>= 1) { if (t < s) red[t] += red[t+s]; __syncthreads(); }
    const float inv = 1.f / red[0];
    __syncthreads();
    const __bf16* q = qb + (long)b * NT_ * D_;
    const int d0 = t, d1 = t + 256;
    float a0 = 0.f, a1 = 0.f;
    for (int n = 0; n < NT_; ++n) {
        const float pn = p[n] * inv;
        a0 += pn * (float)q[(long)n*D_ + d0];
        a1 += pn * (float)q[(long)n*D_ + d1];
    }
    g[b*D_ + d0] = a0;
    g[b*D_ + d1] = a1;
}

// In-place k[b,n,d] *= g[b,d]
__global__ __launch_bounds__(256) void scale_k_kernel(
    __bf16* __restrict__ kb, const float* __restrict__ g)
{
    const long i = (long)blockIdx.x * 256 + threadIdx.x;  // over B*NV*D
    const long per = (long)NV_ * D_;
    const int b = (int)(i / per);
    const int d = (int)(i & (D_-1));
    kb[i] = (__bf16)((float)kb[i] * g[b*D_ + d]);
}

__global__ __launch_bounds__(256) void cast_kernel(
    const float* __restrict__ s, __bf16* __restrict__ d, long n)
{
    const long i = (long)blockIdx.x * 256 + threadIdx.x;
    if (i < n) d[i] = (__bf16)s[i];
}

// verts_f [B,NV,D] fp32 -> x bf16 rows [b*NT + n]
__global__ __launch_bounds__(256) void cast_verts_kernel(
    const float* __restrict__ v, __bf16* __restrict__ xb)
{
    const long i = (long)blockIdx.x * 256 + threadIdx.x;  // over B*NV*D
    const long r = i >> 9;
    const int  d = (int)(i & 511);
    const long b = r / NV_, n = r % NV_;
    xb[((b * NT_ + n) << 9) + d] = (__bf16)v[i];
}

extern "C" void kernel_launch(void* const* d_in, const int* in_sizes, int n_in,
                              void* d_out, int out_size, void* d_ws, size_t ws_size,
                              hipStream_t stream) {
    const float* verts   = (const float*)d_in[0];
    const float* imgf    = (const float*)d_in[1];
    const float* fc_w    = (const float*)d_in[2];
    const float* fc_b    = (const float*)d_in[3];
    const float* q_w     = (const float*)d_in[4];
    const float* q_b     = (const float*)d_in[5];
    const float* k_w     = (const float*)d_in[6];
    const float* k_b     = (const float*)d_in[7];
    const float* w_g     = (const float*)d_in[8];
    const float* proj_w  = (const float*)d_in[9];
    const float* proj_b  = (const float*)d_in[10];
    const float* final_w = (const float*)d_in[11];
    const float* final_b = (const float*)d_in[12];
    (void)in_sizes; (void)n_in; (void)out_size; (void)ws_size;

    char* ws = (char*)d_ws;
    size_t off = 0;
    auto alloc = [&](size_t bytes) -> void* {
        void* p = ws + off;
        off = (off + bytes + 255) & ~(size_t)255;
        return p;
    };

    const long nX   = (long)B_ * NT_ * D_;    // 29,687,808
    const long nImg = (long)B_ * NI_ * ID_;   // 16,777,216
    const long nV   = (long)B_ * NV_ * D_;    // 25,493,504

    __bf16* xb     = (__bf16*)alloc(nX * 2);        // concat(verts, fc(img)) bf16
    __bf16* imb    = (__bf16*)alloc(nImg * 2);      // img_f bf16
    __bf16* qb     = (__bf16*)alloc(nX * 2);        // normalized q bf16
    __bf16* kb     = (__bf16*)alloc(nV * 2);        // normalized k (then g*k) bf16
    __bf16* o1     = (__bf16*)alloc(nV * 2);        // proj output + q bf16
    float*  logits = (float*)alloc((long)B_ * NT_ * 4);
    float*  g      = (float*)alloc((long)B_ * D_ * 4);
    __bf16* fcwb   = (__bf16*)alloc((long)D_ * ID_ * 2);
    __bf16* qwb    = (__bf16*)alloc((long)D_ * D_ * 2);
    __bf16* kwb    = (__bf16*)alloc((long)D_ * D_ * 2);
    __bf16* pwb    = (__bf16*)alloc((long)D_ * D_ * 2);
    __bf16* fwb    = (__bf16*)alloc((long)D_ * D_ * 2);

    const long nWfc = (long)D_ * ID_;
    const long nW   = (long)D_ * D_;
    cast_kernel<<<(nWfc + 255) / 256, 256, 0, stream>>>(fc_w,    fcwb, nWfc);
    cast_kernel<<<(nW   + 255) / 256, 256, 0, stream>>>(q_w,     qwb,  nW);
    cast_kernel<<<(nW   + 255) / 256, 256, 0, stream>>>(k_w,     kwb,  nW);
    cast_kernel<<<(nW   + 255) / 256, 256, 0, stream>>>(proj_w,  pwb,  nW);
    cast_kernel<<<(nW   + 255) / 256, 256, 0, stream>>>(final_w, fwb,  nW);
    cast_kernel<<<(nImg + 255) / 256, 256, 0, stream>>>(imgf,    imb,  nImg);
    cast_verts_kernel<<<nV / 256, 256, 0, stream>>>(verts, xb);

    const dim3 blk(256);
    const float SCALE = 0.08838834764831843f;  // (512/4)^-0.5

    // fc: [B*NI, ID] x [D, ID]^T -> x img rows
    wmma_gemm<0><<<dim3((B_*NI_)/128, 8), blk, 0, stream>>>(
        imb, fcwb, fc_b, nullptr, nullptr, xb, nullptr, ID_);
    // q: [B*NT, D] x [D, D]^T -> qb (raw), then normalize + logits
    wmma_gemm<1><<<dim3((B_*NT_)/128, 8), blk, 0, stream>>>(
        xb, qwb, q_b, nullptr, nullptr, qb, nullptr, D_);
    rownorm_kernel<true><<<B_*NT_, 128, 0, stream>>>(qb, w_g, logits, SCALE);
    // k: vertex rows only
    wmma_gemm<2><<<dim3((B_*NV_)/128, 8), blk, 0, stream>>>(
        xb, kwb, k_b, nullptr, nullptr, kb, nullptr, D_);
    rownorm_kernel<false><<<B_*NV_, 128, 0, stream>>>(kb, nullptr, nullptr, 0.f);
    // softmax over tokens + global query g
    softmax_g_kernel<<<B_, 256, 0, stream>>>(logits, qb, g);
    // k <- g * k (broadcast over tokens)
    scale_k_kernel<<<(int)(nV / 256), 256, 0, stream>>>(kb, g);
    // proj: (g*k) x proj_w^T + proj_b + q -> o1
    wmma_gemm<3><<<dim3((B_*NV_)/128, 8), blk, 0, stream>>>(
        kb, pwb, proj_b, qb, nullptr, o1, nullptr, D_);
    // final: o1 x final_w^T + final_b + verts -> d_out (fp32)
    wmma_gemm<4><<<dim3((B_*NV_)/128, 8), blk, 0, stream>>>(
        o1, fwb, final_b, nullptr, verts, nullptr, (float*)d_out, D_);
}
